// FlashAttention_61237643706834
// MI455X (gfx1250) — matrix operand order repaired and checked by reading, not compiled, not run on hardware
//
#include <hip/hip_runtime.h>
#include <hip/hip_bf16.h>

#define BATCH 2
#define SEQ   4096
#define NTOK  (BATCH * SEQ)   // 8192
#define HEADS 12
#define HD    64
#define DIM   768
#define DQKV  2304
#define FSCALE 0.125f         // 64^-0.5
#define KCHUNK 64
#define NCH   (SEQ / KCHUNK)  // 64

typedef _Float16 h16;
typedef __attribute__((ext_vector_type(16))) _Float16 v16h;
typedef __attribute__((ext_vector_type(8)))  _Float16 v8h;
typedef __attribute__((ext_vector_type(4)))  _Float16 v4h;
typedef __attribute__((ext_vector_type(8)))  float    v8f;
typedef __attribute__((ext_vector_type(4)))  int      i4v;

#if __has_builtin(__builtin_amdgcn_global_load_async_to_lds_b128)
#define HAVE_ASYNC_LDS 1
#endif

// 16-byte global -> LDS copy; async (ASYNCcnt-tracked) when available.
static __device__ __forceinline__ void async_cp16(const h16* g, h16* l) {
#ifdef HAVE_ASYNC_LDS
  __builtin_amdgcn_global_load_async_to_lds_b128(
      (__attribute__((address_space(1))) i4v*)g,
      (__attribute__((address_space(3))) i4v*)l, 0, 0);
#else
  *reinterpret_cast<uint4*>(l) = *reinterpret_cast<const uint4*>(g);
#endif
}

static __device__ __forceinline__ void wait_async0() {
#ifdef HAVE_ASYNC_LDS
#if __has_builtin(__builtin_amdgcn_s_wait_asynccnt)
  __builtin_amdgcn_s_wait_asynccnt(0);
#else
  asm volatile("s_wait_asynccnt 0x0" ::: "memory");
#endif
#endif
}

// ---- WMMA fragment loaders (CDNA5 16x16x32 f16, contiguous k = 16h + s order for A and B, wave32) ----
// A (16x32, MxK): lanes 0-15 hold row=lane, K=0..15;
//                 lanes 16-31 hold row=lane-16, K=16..31.
static __device__ __forceinline__ v16h ld_a(const h16* base, int ld) {
  int lane = threadIdx.x & 31;
  int row  = lane & 15;
  int k0   = (lane < 16) ? 0 : 16;
  return *reinterpret_cast<const v16h*>(base + row * ld + k0);
}

// B (32x16, KxN): lane holds column n=lane%16; lanes 0-15 K=0..15, lanes 16-31 K=16..31.
// baseT is the transposed matrix: row n holds K contiguously.
static __device__ __forceinline__ v16h ld_b(const h16* baseT, int ld) {
  int lane = threadIdx.x & 31;
  int n    = lane & 15;
  int k0   = (lane < 16) ? 0 : 16;
  return *reinterpret_cast<const v16h*>(baseT + n * ld + k0);
}

static __device__ __forceinline__ v8f wmma_f16(v16h a, v16h b, v8f c) {
  return __builtin_amdgcn_wmma_f32_16x16x32_f16(false, a, false, b, (short)0, c, false, false);
}

// ---- prep kernels ----
__global__ void k_cvt(const float* __restrict__ in, h16* __restrict__ out, int n4) {
  int i = blockIdx.x * blockDim.x + threadIdx.x;
  if (i < n4) {
    float4 f = reinterpret_cast<const float4*>(in)[i];
    v4h o;
    o[0] = (h16)f.x; o[1] = (h16)f.y; o[2] = (h16)f.z; o[3] = (h16)f.w;
    reinterpret_cast<v4h*>(out)[i] = o;
  }
}

__global__ void k_transpose_cvt(const float* __restrict__ in, h16* __restrict__ out,
                                int rows, int cols) {
  int idx = blockIdx.x * blockDim.x + threadIdx.x;
  if (idx >= rows * cols) return;
  int r = idx / cols, c = idx % cols;
  out[(size_t)c * rows + r] = (h16)in[idx];
}

// ---- QKV projection: [8192,768] x [768,2304] -> scatter into Q,K (b,h,n,d) and V^T (b,h,d,n)
__global__ void __launch_bounds__(128) k_qkv(const h16* __restrict__ xh,
                                             const h16* __restrict__ wT,
                                             h16* __restrict__ Qf,
                                             h16* __restrict__ Kf,
                                             h16* __restrict__ Vt) {
  int wave = threadIdx.x >> 5;
  int lane = threadIdx.x & 31;
  int m0 = blockIdx.x * 64;
  int n0 = (blockIdx.y * 4 + wave) * 64;   // multiple of 64 -> one (s,h) per wave
  v8f acc[4][4] = {};
  for (int k0 = 0; k0 < DIM; k0 += 32) {
    v16h a[4], bfr[4];
#pragma unroll
    for (int i = 0; i < 4; ++i) a[i]   = ld_a(xh + (size_t)(m0 + 16 * i) * DIM + k0, DIM);
#pragma unroll
    for (int j = 0; j < 4; ++j) bfr[j] = ld_b(wT + (size_t)(n0 + 16 * j) * DIM + k0, DIM);
#pragma unroll
    for (int i = 0; i < 4; ++i)
#pragma unroll
      for (int j = 0; j < 4; ++j) acc[i][j] = wmma_f16(a[i], bfr[j], acc[i][j]);
  }
  int s  = n0 / DIM;             // 0:Q 1:K 2:V (uniform per wave)
  int hh = (n0 % DIM) / HD;      // head
#pragma unroll
  for (int i = 0; i < 4; ++i)
#pragma unroll
    for (int j = 0; j < 4; ++j)
#pragma unroll
      for (int r = 0; r < 8; ++r) {
        int mrow = m0 + 16 * i + r + ((lane < 16) ? 0 : 8);
        int d    = 16 * j + (lane & 15);
        int bb   = mrow >> 12;
        int q    = mrow & 4095;
        float v  = acc[i][j][r];
        size_t bhn = ((size_t)(bb * HEADS + hh) * SEQ + q) * HD + d;
        if (s == 0)      Qf[bhn] = (h16)(v * FSCALE);
        else if (s == 1) Kf[bhn] = (h16)v;
        else             Vt[((size_t)(bb * HEADS + hh) * HD + d) * SEQ + q] = (h16)v;
      }
}

// ---- Flash attention: block = 4 waves = 64 queries of one (b,h) ----
// Double-buffered async K/V staging, 64-key chunks.
__global__ void __launch_bounds__(128) k_attn(const h16* __restrict__ Qf,
                                              const h16* __restrict__ Kf,
                                              const h16* __restrict__ Vt,
                                              h16* __restrict__ Ao) {
  __shared__ __align__(32) h16 Kt[2][KCHUNK * HD];    // [key][hd]   8KB x2
  __shared__ __align__(32) h16 Vl[2][HD * KCHUNK];    // [hd][kv]    8KB x2
  __shared__ __align__(32) h16 Pl[4][16 * KCHUNK];    // per-wave P  2KB x4

  int bid  = blockIdx.x;
  int qt   = bid & 63;        // SEQ/64 tiles
  int bh   = bid >> 6;        // 0..23
  int b    = bh / HEADS, h = bh % HEADS;
  int wave = threadIdx.x >> 5;
  int lane = threadIdx.x & 31;
  int q0   = qt * 64 + wave * 16;

  const h16* Qbh = Qf + (size_t)bh * SEQ * HD;
  const h16* Kbh = Kf + (size_t)bh * SEQ * HD;
  const h16* Vbh = Vt + (size_t)bh * HD * SEQ;

  v16h aq0 = ld_a(Qbh + (size_t)q0 * HD, HD);        // hd 0..31
  v16h aq1 = ld_a(Qbh + (size_t)q0 * HD + 32, HD);   // hd 32..63

  v8f acc[4] = {};
  float mrow[8], lrow[8];
#pragma unroll
  for (int r = 0; r < 8; ++r) { mrow[r] = -1e30f; lrow[r] = 0.0f; }

  auto issue = [&](int buf, int kv0) {
    // K tile: 64x64 halves = 8KB contiguous = 512 x 16B
#pragma unroll
    for (int c = 0; c < 4; ++c) {
      int idx = threadIdx.x + 128 * c;
      async_cp16(Kbh + (size_t)kv0 * HD + idx * 8, &Kt[buf][idx * 8]);
    }
    // V^T tile: 64 hd-rows x 64 kv halves (128B per row)
#pragma unroll
    for (int c = 0; c < 4; ++c) {
      int idx = threadIdx.x + 128 * c;
      int d = idx >> 3, cc = idx & 7;
      async_cp16(Vbh + (size_t)d * SEQ + kv0 + cc * 8, &Vl[buf][idx * 8]);
    }
  };

  issue(0, 0);
  for (int i = 0; i < NCH; ++i) {
    int buf = i & 1;
    wait_async0();      // my wave's pending tile copies (incl. buf) are in LDS
    __syncthreads();    // everyone's copies landed; prev readers of buf^1 done
    if (i + 1 < NCH) issue(buf ^ 1, (i + 1) * KCHUNK);  // overlap with compute

    const h16* Kc = Kt[buf];
    const h16* Vc = Vl[buf];

    // S = Q K^T : four 16x16 key tiles, hd reduced in two K=32 steps each
    v8f z = {};
    v8f s[4];
#pragma unroll
    for (int t = 0; t < 4; ++t) {
      s[t] = wmma_f16(aq0, ld_b(Kc + 16 * t * HD, HD), z);
      s[t] = wmma_f16(aq1, ld_b(Kc + 16 * t * HD + 32, HD), s[t]);
    }

    // online softmax over 64-key chunk (rows live in 16-lane groups)
    float cm[8], rs[8];
#pragma unroll
    for (int r = 0; r < 8; ++r)
      cm[r] = fmaxf(fmaxf(s[0][r], s[1][r]), fmaxf(s[2][r], s[3][r]));
#pragma unroll
    for (int off = 1; off < 16; off <<= 1)
#pragma unroll
      for (int r = 0; r < 8; ++r) cm[r] = fmaxf(cm[r], __shfl_xor(cm[r], off));
#pragma unroll
    for (int r = 0; r < 8; ++r) {
      float mn = fmaxf(mrow[r], cm[r]);
      float al = __expf(mrow[r] - mn);
      mrow[r] = mn;
      lrow[r] *= al;
#pragma unroll
      for (int t = 0; t < 4; ++t) acc[t][r] *= al;
      float sum = 0.0f;
#pragma unroll
      for (int t = 0; t < 4; ++t) { s[t][r] = __expf(s[t][r] - mn); sum += s[t][r]; }
      rs[r] = sum;
    }
#pragma unroll
    for (int off = 1; off < 16; off <<= 1)
#pragma unroll
      for (int r = 0; r < 8; ++r) rs[r] += __shfl_xor(rs[r], off);
#pragma unroll
    for (int r = 0; r < 8; ++r) lrow[r] += rs[r];

    // D-layout f32 -> A-layout f16 via per-wave LDS round-trip
    h16* P = Pl[wave];
    {
      int roff = (lane < 16) ? 0 : 8;
      int col  = lane & 15;
#pragma unroll
      for (int t = 0; t < 4; ++t)
#pragma unroll
        for (int r = 0; r < 8; ++r)
          P[(r + roff) * KCHUNK + 16 * t + col] = (h16)s[t][r];
    }
    __syncthreads();
    v16h pf0 = ld_a(P, KCHUNK);        // kv 0..31
    v16h pf1 = ld_a(P + 32, KCHUNK);   // kv 32..63

    // O += P @ V  (4 hd sub-tiles, two K=32 steps each)
#pragma unroll
    for (int j = 0; j < 4; ++j) {
      acc[j] = wmma_f16(pf0, ld_b(Vc + 16 * j * KCHUNK, KCHUNK), acc[j]);
      acc[j] = wmma_f16(pf1, ld_b(Vc + 16 * j * KCHUNK + 32, KCHUNK), acc[j]);
    }
  }

  // normalize and write [b, q, h*64+d] as f16
#pragma unroll
  for (int r = 0; r < 8; ++r) lrow[r] = 1.0f / lrow[r];
#pragma unroll
  for (int j = 0; j < 4; ++j)
#pragma unroll
    for (int r = 0; r < 8; ++r) {
      int q = q0 + r + ((lane < 16) ? 0 : 8);
      int d = 16 * j + (lane & 15);
      Ao[((size_t)(b * SEQ) + q) * DIM + h * HD + d] = (h16)(acc[j][r] * lrow[r]);
    }
}

// ---- output projection: [8192,768] x [768,768] + bias -> f32 out ----
__global__ void __launch_bounds__(128) k_proj(const h16* __restrict__ A,
                                              const h16* __restrict__ wT,
                                              const float* __restrict__ bias,
                                              float* __restrict__ out) {
  int wave = threadIdx.x >> 5;
  int lane = threadIdx.x & 31;
  int m0 = blockIdx.x * 64;
  int n0 = (blockIdx.y * 4 + wave) * 64;
  v8f acc[4][4] = {};
  for (int k0 = 0; k0 < DIM; k0 += 32) {
    v16h a[4], bfr[4];
#pragma unroll
    for (int i = 0; i < 4; ++i) a[i]   = ld_a(A + (size_t)(m0 + 16 * i) * DIM + k0, DIM);
#pragma unroll
    for (int j = 0; j < 4; ++j) bfr[j] = ld_b(wT + (size_t)(n0 + 16 * j) * DIM + k0, DIM);
#pragma unroll
    for (int i = 0; i < 4; ++i)
#pragma unroll
      for (int j = 0; j < 4; ++j) acc[i][j] = wmma_f16(a[i], bfr[j], acc[i][j]);
  }
#pragma unroll
  for (int i = 0; i < 4; ++i)
#pragma unroll
    for (int j = 0; j < 4; ++j)
#pragma unroll
      for (int r = 0; r < 8; ++r) {
        int mrow = m0 + 16 * i + r + ((lane < 16) ? 0 : 8);
        int n    = n0 + 16 * j + (lane & 15);
        out[(size_t)mrow * DIM + n] = acc[i][j][r] + bias[n];
      }
}

extern "C" void kernel_launch(void* const* d_in, const int* in_sizes, int n_in,
                              void* d_out, int out_size, void* d_ws, size_t ws_size,
                              hipStream_t stream) {
  (void)in_sizes; (void)n_in; (void)out_size; (void)ws_size;
  const float* x      = (const float*)d_in[0];
  const float* w_qkv  = (const float*)d_in[1];
  const float* w_proj = (const float*)d_in[2];
  const float* b_proj = (const float*)d_in[3];
  float* out = (float*)d_out;

  char* ws = (char*)d_ws;
  size_t off = 0;
  auto alloc = [&](size_t bytes) -> void* {
    void* p = ws + off;
    off = (off + bytes + 255) & ~(size_t)255;
    return p;
  };
  h16* xh     = (h16*)alloc((size_t)NTOK * DIM * sizeof(h16));
  h16* wqkvT  = (h16*)alloc((size_t)DQKV * DIM * sizeof(h16));
  h16* wprojT = (h16*)alloc((size_t)DIM * DIM * sizeof(h16));
  h16* Qf     = (h16*)alloc((size_t)BATCH * HEADS * SEQ * HD * sizeof(h16));
  h16* Kf     = (h16*)alloc((size_t)BATCH * HEADS * SEQ * HD * sizeof(h16));
  h16* Vt     = (h16*)alloc((size_t)BATCH * HEADS * HD * SEQ * sizeof(h16));
  h16* Ao     = (h16*)alloc((size_t)NTOK * DIM * sizeof(h16));

  k_cvt<<<(NTOK * DIM / 4 + 255) / 256, 256, 0, stream>>>(x, xh, NTOK * DIM / 4);
  k_transpose_cvt<<<(DIM * DQKV + 255) / 256, 256, 0, stream>>>(w_qkv, wqkvT, DIM, DQKV);
  k_transpose_cvt<<<(DIM * DIM + 255) / 256, 256, 0, stream>>>(w_proj, wprojT, DIM, DIM);

  k_qkv<<<dim3(NTOK / 64, DQKV / 256), 128, 0, stream>>>(xh, wqkvT, Qf, Kf, Vt);
  k_attn<<<BATCH * HEADS * (SEQ / 64), 128, 0, stream>>>(Qf, Kf, Vt, Ao);
  k_proj<<<dim3(NTOK / 64, DIM / 256), 128, 0, stream>>>(Ao, wprojT, b_proj, out);
}
